// AttentionBlock_21397527068789
// MI455X (gfx1250) — matrix operand order repaired and checked by reading, not compiled, not run on hardware
//
#include <hip/hip_runtime.h>
#include <hip/hip_bf16.h>

// AttentionBlock for MI455X (gfx1250, wave32, WMMA).
// Pipeline (all f16 data / f32 accumulate):
//   1. cvt_w:    wq/wk/wv/wp f32 -> f16
//   2. gn:       GroupNorm(x) -> Xt [B][N][C] f16 (transposed for NT GEMMs)
//   3. gemm_nt:  Qt = Xt*Wq^T + bq   [B][N][C]
//   4. gemm_nt:  Kt = Xt*Wk^T + bk   [B][N][C]
//   5. gemm_nt:  V  = Wv*Xt^T + bv   [B][C][N]
//   6. attn:     flash attention, 16-row i-tile per wave -> Ot [B][N][C]
//   7. gemm_nt:  out = Wp*Ot^T + bp + x   (f32, residual fused)
// Workspace: 4*65536*2 + 4*(8*4096*256*2) = 67,633,152 bytes (Ot aliases Xt).

typedef __attribute__((ext_vector_type(16))) _Float16 v16h;
typedef __attribute__((ext_vector_type(8)))  _Float16 v8h;
typedef __attribute__((ext_vector_type(8)))  float    v8f;

#define C_DIM 256
#define N_SP  4096
#define B_DIM 8

// ---------------- WMMA helpers ----------------

static __device__ __forceinline__ v8f wmma_f16(v16h a, v16h b, v8f c) {
  // D(16x16,f32) = A(16x32,f16) * B(32x16,f16) + C
  return __builtin_amdgcn_wmma_f32_16x16x32_f16(
      /*neg_a=*/false, a, /*neg_b=*/false, b,
      /*c_mod=*/(short)0, c, /*reuse_a=*/false, /*reuse_b=*/false);
}

// A fragment (16x32 f16, M rows of a row-major [rows][ld] matrix, K chunk k0..k0+31).
// ISA layout: lanes 0-15 hold row M=lane, K = k0+0..7 and k0+16..23;
//             lanes 16-31 hold row M=lane-16, K = k0+8..15 and k0+24..31.
static __device__ __forceinline__ v16h frag_a(const _Float16* p, int ld, int row0, int k0) {
  int lane = threadIdx.x & 31;
  const _Float16* base = p + (size_t)(row0 + (lane & 15)) * ld + k0 + ((lane & 16) ? 8 : 0);
  v8h lo = *(const v8h*)base;          // K + {0..7}  (or {8..15})
  v8h hi = *(const v8h*)(base + 16);   // K + {16..23}(or {24..31})
  v16h r;
#pragma unroll
  for (int i = 0; i < 8; ++i) { r[i] = lo[i]; r[i + 8] = hi[i]; }
  return r;
}

// B fragment (32x16 f16). Column n comes from ROW (row0+n) of a row-major [rows][ld]
// matrix (i.e. B^T stored row-major -> NT GEMM).
// ISA layout: lanes 0-15 hold N=lane, K = k0+0..7 and k0+16..23;
//             lanes 16-31 hold N=lane-16, K = k0+8..15 and k0+24..31.
static __device__ __forceinline__ v16h frag_b(const _Float16* p, int ld, int row0, int k0) {
  int lane = threadIdx.x & 31;
  const _Float16* base = p + (size_t)(row0 + (lane & 15)) * ld + k0 + ((lane & 16) ? 8 : 0);
  v8h lo = *(const v8h*)base;
  v8h hi = *(const v8h*)(base + 16);
  v16h r;
#pragma unroll
  for (int i = 0; i < 8; ++i) { r[i] = lo[i]; r[i + 8] = hi[i]; }
  return r;
}

// Reductions within a 16-lane half of a wave32 (rows of a WMMA D tile live in
// 16-lane halves; xor masks < 16 never cross the half boundary).
static __device__ __forceinline__ float half_max(float v) {
#pragma unroll
  for (int m = 1; m < 16; m <<= 1) v = fmaxf(v, __shfl_xor(v, m, 32));
  return v;
}
static __device__ __forceinline__ float half_sum(float v) {
#pragma unroll
  for (int m = 1; m < 16; m <<= 1) v += __shfl_xor(v, m, 32);
  return v;
}

// ---------------- 1. weight conversion ----------------

__global__ void cvt_w_kernel(const float* __restrict__ wq, const float* __restrict__ wk,
                             const float* __restrict__ wv, const float* __restrict__ wp,
                             _Float16* __restrict__ w16) {
  int i = blockIdx.x * blockDim.x + threadIdx.x;
  if (i < C_DIM * C_DIM) {
    w16[i]                       = (_Float16)wq[i];
    w16[C_DIM * C_DIM + i]       = (_Float16)wk[i];
    w16[2 * C_DIM * C_DIM + i]   = (_Float16)wv[i];
    w16[3 * C_DIM * C_DIM + i]   = (_Float16)wp[i];
  }
}

// ---------------- 2. GroupNorm -> Xt [B][N][C] f16 ----------------

__global__ __launch_bounds__(256) void gn_kernel(const float* __restrict__ x,
                                                 const float* __restrict__ gamma,
                                                 const float* __restrict__ beta,
                                                 _Float16* __restrict__ Xt) {
  const int CPG = 32;                       // channels per group (256/8)
  const int b = blockIdx.x >> 3;
  const int g = blockIdx.x & 7;
  const float* xb = x + ((size_t)b * C_DIM + g * CPG) * N_SP;
  const int cnt = CPG * N_SP;               // 131072

  float s = 0.f, ss = 0.f;
  for (int e = threadIdx.x; e < cnt; e += 256) {
    float v = xb[e];
    s += v; ss += v * v;
  }
#pragma unroll
  for (int m = 1; m < 32; m <<= 1) { s += __shfl_xor(s, m, 32); ss += __shfl_xor(ss, m, 32); }

  __shared__ float rs[8], rss[8];
  __shared__ float s_mu, s_inv;
  int wv = threadIdx.x >> 5, lane = threadIdx.x & 31;
  if (lane == 0) { rs[wv] = s; rss[wv] = ss; }
  __syncthreads();
  if (threadIdx.x == 0) {
    float ts = 0.f, tss = 0.f;
    for (int i = 0; i < 8; ++i) { ts += rs[i]; tss += rss[i]; }
    float mu = ts / (float)cnt;
    float var = tss / (float)cnt - mu * mu;
    s_mu = mu;
    s_inv = rsqrtf(var + 1e-5f);
  }
  __syncthreads();
  float mu = s_mu, inv = s_inv;

  for (int e = threadIdx.x; e < cnt; e += 256) {
    int cl = e >> 12;                       // e / 4096
    int n  = e & (N_SP - 1);
    int c  = g * CPG + cl;
    float v = (xb[e] - mu) * inv * gamma[c] + beta[c];
    Xt[((size_t)b * N_SP + n) * C_DIM + c] = (_Float16)v;
  }
}

// ---------------- 3/4/5/7. NT GEMM: D[m][n] = sum_k A[m][k]*B[n][k] + bias ----------------
// 8 waves/block, each wave computes a 32x32 tile (2x2 WMMA accumulators).

__global__ __launch_bounds__(256) void gemm_nt_kernel(
    const _Float16* __restrict__ A, const _Float16* __restrict__ Bm,
    _Float16* __restrict__ outH, float* __restrict__ outF,
    const float* __restrict__ res, const float* __restrict__ bias,
    int M, int N, int K, int lda, int ldb, int ldo, int biasOnM,
    unsigned long long sA, unsigned long long sB,
    unsigned long long sO, unsigned long long sR, int Ntiles) {
  int wv   = threadIdx.x >> 5;
  int tile = blockIdx.x * 8 + wv;
  int Mt   = M >> 5;
  if (tile >= Mt * Ntiles) return;          // wave-uniform guard (EXEC stays all-1)
  int mt = tile / Ntiles, nt = tile % Ntiles;
  int m0 = mt << 5, n0 = nt << 5;

  const _Float16* Ab = A  + (size_t)blockIdx.y * sA;
  const _Float16* Bb = Bm + (size_t)blockIdx.y * sB;

  v8f z = {};
  v8f acc00 = z, acc01 = z, acc10 = z, acc11 = z;

  for (int k = 0; k < K; k += 32) {
    v16h a0 = frag_a(Ab, lda, m0,      k);
    v16h a1 = frag_a(Ab, lda, m0 + 16, k);
    v16h b0 = frag_b(Bb, ldb, n0,      k);
    v16h b1 = frag_b(Bb, ldb, n0 + 16, k);
    acc00 = wmma_f16(a0, b0, acc00);
    acc01 = wmma_f16(a0, b1, acc01);
    acc10 = wmma_f16(a1, b0, acc10);
    acc11 = wmma_f16(a1, b1, acc11);
  }

  int lane = threadIdx.x & 31;
  int half = lane >> 4, jl = lane & 15;
  v8f accs[2][2] = {{acc00, acc01}, {acc10, acc11}};
#pragma unroll
  for (int mi = 0; mi < 2; ++mi)
#pragma unroll
    for (int ni = 0; ni < 2; ++ni)
#pragma unroll
      for (int r = 0; r < 8; ++r) {
        int m = m0 + 16 * mi + r + 8 * half;
        int n = n0 + 16 * ni + jl;
        float v = accs[mi][ni][r] + (biasOnM ? bias[m] : bias[n]);
        size_t o = (size_t)m * ldo + n;
        if (outF) {
          float rv = res ? res[(size_t)blockIdx.y * sR + o] : 0.f;
          outF[(size_t)blockIdx.y * sO + o] = v + rv;
        } else {
          outH[(size_t)blockIdx.y * sO + o] = (_Float16)v;
        }
      }
}

// ---------------- 6. Flash attention ----------------
// One wave owns a 16-row i-tile. S = Qt_tile * Kt^T (K=256, 8 WMMA per 16-j tile),
// online softmax in f32, P bounced through padded LDS to A-layout,
// O(16 x 256) accumulated in 16 f32 WMMA tiles (128 VGPRs).

#define ATT_WAVES 4
#define QLD 264   // 256 + 8 f16 pad -> conflict-free A-frag reads
#define PLD 40    // 32 + 8 f16 pad

__global__ __launch_bounds__(128, 1) void attn_kernel(
    const _Float16* __restrict__ Qt, const _Float16* __restrict__ Kt,
    const _Float16* __restrict__ V, _Float16* __restrict__ Ot) {
  __shared__ _Float16 qtile_s[ATT_WAVES][16 * QLD];
  __shared__ _Float16 ptile_s[ATT_WAVES][16 * PLD];

  int wv   = threadIdx.x >> 5;
  int lane = threadIdx.x & 31;
  int gw   = blockIdx.x * ATT_WAVES + wv;    // 2048 waves total
  int b    = gw >> 8;                        // 256 i-tiles per batch
  int i0   = (gw & 255) << 4;

  const _Float16* Qb = Qt + (size_t)b * N_SP * C_DIM;
  const _Float16* Kb = Kt + (size_t)b * N_SP * C_DIM;
  const _Float16* Vb = V  + (size_t)b * C_DIM * N_SP;
  _Float16*       Ob = Ot + (size_t)b * N_SP * C_DIM;

  _Float16* qs = qtile_s[wv];
  _Float16* ps = ptile_s[wv];

  // Stage the 16x256 Q tile into LDS (u32 copies; wave-private region).
  for (int e = lane; e < 16 * 128; e += 32) {
    int row = e >> 7, col = e & 127;
    ((unsigned int*)(qs + row * QLD))[col] =
        ((const unsigned int*)(Qb + (size_t)(i0 + row) * C_DIM))[col];
  }

  v8f o[16];
#pragma unroll
  for (int t = 0; t < 16; ++t) { v8f z = {}; o[t] = z; }
  float mrow[8], lrow[8];
#pragma unroll
  for (int r = 0; r < 8; ++r) { mrow[r] = -1e30f; lrow[r] = 0.f; }

  const float scale = 0.0625f;               // 1/sqrt(256)
  int half = lane >> 4, jl = lane & 15;

  for (int j0 = 0; j0 < N_SP; j0 += 32) {
    if (j0 + 32 < N_SP)                       // pull next K rows toward L2/L0
      __builtin_prefetch(Kb + (size_t)(j0 + 32 + lane) * C_DIM, 0, 1);

    // ---- S tiles: rows i0..i0+15, cols j0..j0+15 (s0) and j0+16..j0+31 (s1)
    v8f s0 = {}, s1 = {};
#pragma unroll
    for (int kc = 0; kc < 8; ++kc) {
      v16h a  = frag_a(qs, QLD, 0, kc * 32);
      v16h k0 = frag_b(Kb, C_DIM, j0,      kc * 32);
      v16h k1 = frag_b(Kb, C_DIM, j0 + 16, kc * 32);
      s0 = wmma_f16(a, k0, s0);
      s1 = wmma_f16(a, k1, s1);
    }

    // ---- online softmax (row r lives in a 16-lane half; lanes>=16 carry rows r+8)
    float fac[8];
#pragma unroll
    for (int r = 0; r < 8; ++r) {
      float v0 = s0[r] * scale, v1 = s1[r] * scale;
      float mc = half_max(fmaxf(v0, v1));
      float mn = fmaxf(mrow[r], mc);
      float cf = __expf(mrow[r] - mn);
      float p0 = __expf(v0 - mn), p1 = __expf(v1 - mn);
      lrow[r] = lrow[r] * cf + half_sum(p0 + p1);
      mrow[r] = mn;
      fac[r]  = cf;
      int row = r + 8 * half;                 // local i index of this lane's value
      ps[row * PLD + jl]      = (_Float16)p0;
      ps[row * PLD + 16 + jl] = (_Float16)p1;
    }

    // ---- rescale O accumulators (tile M=i matches stats indexing exactly)
#pragma unroll
    for (int t = 0; t < 16; ++t)
#pragma unroll
      for (int r = 0; r < 8; ++r) o[t][r] *= fac[r];

    // ---- O += P(16x32) * V(32xC): one A-frag, 16 B-frags / WMMAs
    v16h pf = frag_a(ps, PLD, 0, 0);
#pragma unroll
    for (int t = 0; t < 16; ++t) {
      v16h vb = frag_b(Vb, N_SP, t * 16, j0); // "rows" of B are channels c
      o[t] = wmma_f16(pf, vb, o[t]);
    }
  }

  // ---- normalize and store Ot [N][C] f16
  float inv[8];
#pragma unroll
  for (int r = 0; r < 8; ++r) inv[r] = 1.0f / lrow[r];
#pragma unroll
  for (int t = 0; t < 16; ++t)
#pragma unroll
    for (int r = 0; r < 8; ++r) {
      int i = i0 + r + 8 * half;
      int c = t * 16 + jl;
      Ob[(size_t)i * C_DIM + c] = (_Float16)(o[t][r] * inv[r]);
    }
}

// ---------------- host launcher ----------------

extern "C" void kernel_launch(void* const* d_in, const int* in_sizes, int n_in,
                              void* d_out, int out_size, void* d_ws, size_t ws_size,
                              hipStream_t stream) {
  (void)in_sizes; (void)n_in; (void)out_size; (void)ws_size;
  const float* x     = (const float*)d_in[0];
  const float* gamma = (const float*)d_in[1];
  const float* beta  = (const float*)d_in[2];
  const float* wq    = (const float*)d_in[3];
  const float* bq    = (const float*)d_in[4];
  const float* wk    = (const float*)d_in[5];
  const float* bk    = (const float*)d_in[6];
  const float* wvp   = (const float*)d_in[7];
  const float* bv    = (const float*)d_in[8];
  const float* wp    = (const float*)d_in[9];
  const float* bp    = (const float*)d_in[10];
  float* out = (float*)d_out;

  _Float16* w16  = (_Float16*)d_ws;
  _Float16* wq16 = w16;
  _Float16* wk16 = w16 + C_DIM * C_DIM;
  _Float16* wv16 = w16 + 2 * C_DIM * C_DIM;
  _Float16* wp16 = w16 + 3 * C_DIM * C_DIM;
  _Float16* base = (_Float16*)((char*)d_ws + (size_t)4 * C_DIM * C_DIM * 2);
  const size_t per = (size_t)B_DIM * N_SP * C_DIM;   // 8M f16 elems per tensor
  _Float16* Xt = base;
  _Float16* Qt = base + per;
  _Float16* Kt = base + 2 * per;
  _Float16* Vm = base + 3 * per;
  _Float16* Ot = Xt;   // Xt dead after V GEMM -> alias

  const unsigned long long sNC = (unsigned long long)N_SP * C_DIM;

  cvt_w_kernel<<<256, 256, 0, stream>>>(wq, wk, wvp, wp, w16);
  gn_kernel<<<B_DIM * 8, 256, 0, stream>>>(x, gamma, beta, Xt);

  // Qt[b] = Xt[b] * Wq^T + bq : M=4096 (spatial), N=256 (out ch), bias on N
  gemm_nt_kernel<<<dim3(128, B_DIM), 256, 0, stream>>>(
      Xt, wq16, Qt, nullptr, nullptr, bq,
      N_SP, C_DIM, C_DIM, C_DIM, C_DIM, C_DIM, /*biasOnM=*/0,
      sNC, 0ull, sNC, 0ull, /*Ntiles=*/C_DIM / 32);
  // Kt
  gemm_nt_kernel<<<dim3(128, B_DIM), 256, 0, stream>>>(
      Xt, wk16, Kt, nullptr, nullptr, bk,
      N_SP, C_DIM, C_DIM, C_DIM, C_DIM, C_DIM, 0,
      sNC, 0ull, sNC, 0ull, C_DIM / 32);
  // V[b] = Wv * Xt[b]^T + bv : M=256 (out ch), N=4096 (spatial), bias on M
  gemm_nt_kernel<<<dim3(128, B_DIM), 256, 0, stream>>>(
      wv16, Xt, Vm, nullptr, nullptr, bv,
      C_DIM, N_SP, C_DIM, C_DIM, C_DIM, N_SP, /*biasOnM=*/1,
      0ull, sNC, sNC, 0ull, N_SP / 32);

  attn_kernel<<<(B_DIM * (N_SP / 16)) / ATT_WAVES, 128, 0, stream>>>(Qt, Kt, Vm, Ot);

  // out[b] = Wp * Ot[b]^T + bp + x[b] : f32 output with fused residual
  gemm_nt_kernel<<<dim3(128, B_DIM), 256, 0, stream>>>(
      wp16, Ot, nullptr, out, x, bp,
      C_DIM, N_SP, C_DIM, C_DIM, C_DIM, N_SP, 1,
      0ull, sNC, sNC, sNC, N_SP / 32);
}
